// SelfAttention_1606317769257
// MI455X (gfx1250) — matrix operand order repaired and checked by reading, not compiled, not run on hardware
//
#include <hip/hip_runtime.h>
#include <cstdint>

typedef __attribute__((ext_vector_type(16))) _Float16 v16h;
typedef __attribute__((ext_vector_type(8)))  _Float16 v8h;
typedef __attribute__((ext_vector_type(8)))  float    v8f;

#define BB 8
#define CC 256
#define C8V 32
#define NN 4096
#define ETS 80   // LDS E' row stride in halves: rows 16B-aligned, banks spread
#define LOG2E 1.4426950408889634f

static __device__ __forceinline__ v16h cat8(v8h lo, v8h hi) {
    return __builtin_shufflevector(lo, hi, 0, 1, 2, 3, 4, 5, 6, 7,
                                           8, 9, 10, 11, 12, 13, 14, 15);
}
// A-fragment (16x32 f16): lane holds row m=l%16; K = h + 16*half (contiguous).
// rowp = base of this lane's 32 contiguous K f16.
static __device__ __forceinline__ v16h load_fragA(const _Float16* rowp, int half) {
    const v8h* p = (const v8h*)rowp;
    return cat8(p[2 * half], p[2 * half + 1]);
}
// B-fragment (32x16 f16): lane holds col n=l%16; K = h + 16*half (contiguous).
static __device__ __forceinline__ v16h load_fragB(const _Float16* colp, int half) {
    const v8h* p = (const v8h*)colp;
    return cat8(p[2 * half], p[2 * half + 1]);
}
static __device__ __forceinline__ v8f wmma_f16(v16h a, v16h b, v8f c) {
    return __builtin_amdgcn_wmma_f32_16x16x32_f16(false, a, false, b,
                                                  (short)0, c, false, false);
}
static __device__ __forceinline__ float exp2fast(float x) {
    return __builtin_amdgcn_exp2f(x);                 // v_exp_f32
}

// ---------------------------------------------------------------------------
// One-time converts: x (B,C,N) f32 -> xT (B,N,C) f16 via LDS tile transpose.
// ---------------------------------------------------------------------------
__global__ void __launch_bounds__(256) x_to_f16T(const float* __restrict__ x,
                                                 _Float16* __restrict__ xT)
{
    __shared__ _Float16 t[32][72];
    const int b  = blockIdx.z;
    const int c0 = blockIdx.y * 32;
    const int n0 = blockIdx.x * 64;
    const int tid = threadIdx.x;
    const int nl = tid & 63, cq = tid >> 6;
    #pragma unroll
    for (int r = 0; r < 8; ++r) {             // coalesced 64-wide dword loads
        const int c = r * 4 + cq;
        t[c][nl] = (_Float16)x[((size_t)b * CC + c0 + c) * NN + n0 + nl];
    }
    __syncthreads();
    const int nr = tid >> 2, cs = (tid & 3) * 8;
    v8h v;
    #pragma unroll
    for (int j = 0; j < 8; ++j) v[j] = t[cs + j][nr];
    *(v8h*)(xT + ((size_t)b * NN + n0 + nr) * CC + c0 + cs) = v;  // b128 store
}

__global__ void f32_to_f16(const float* __restrict__ s, _Float16* __restrict__ d, int n)
{
    const int i = blockIdx.x * 256 + threadIdx.x;
    if (i < n) d[i] = (_Float16)s[i];
}

// ---------------------------------------------------------------------------
// Projection GEMM: out[oc,n] = scale*(sum_c W[oc,c]*x[b,c,n] + bias[oc]).
// A = W f16 rows (contiguous K), B = xT rows (contiguous K); 2x b128 / frag.
// WRITET=1 -> (B,N,OC) f16 (b128 row stores, for f/g); 0 -> (B,OC,N) (for hh).
// scale = log2(e) for f (so softmax runs in exp2 domain), 1.0 otherwise.
// ---------------------------------------------------------------------------
template <int WRITET>
__global__ void __launch_bounds__(256) proj_gemm(
    const _Float16* __restrict__ W, const float* __restrict__ bias,
    const _Float16* __restrict__ xT, int OC, float scale,
    _Float16* __restrict__ outp)
{
    const int tiles_oc = OC / 16, tiles_n = NN / 16;
    const int tile = blockIdx.x * 8 + (threadIdx.x >> 5);
    if (tile >= BB * tiles_oc * tiles_n) return;          // wave-uniform
    const int b   = tile / (tiles_oc * tiles_n);
    const int r   = tile % (tiles_oc * tiles_n);
    const int oc0 = (r / tiles_n) * 16;
    const int n0  = (r % tiles_n) * 16;
    const int l = threadIdx.x & 31, col = l & 15, half = l >> 4;

    v8f acc = *(const v8f*)(bias + oc0 + 8 * half);
    const _Float16* wrow = W  + (size_t)(oc0 + col) * CC;
    const _Float16* xrow = xT + ((size_t)b * NN + n0 + col) * CC;
    #pragma unroll
    for (int c0 = 0; c0 < CC; c0 += 32)
        acc = wmma_f16(load_fragA(wrow + c0, half), load_fragB(xrow + c0, half), acc);

    if (WRITET) {
        v8h o;
        #pragma unroll
        for (int v = 0; v < 8; ++v) o[v] = (_Float16)(acc[v] * scale);
        *(v8h*)(outp + ((size_t)b * NN + n0 + col) * OC + oc0 + 8 * half) = o;
    } else {
        _Float16* orow = outp + ((size_t)b * OC + oc0 + 8 * half) * NN + n0 + col;
        #pragma unroll
        for (int v = 0; v < 8; ++v) orow[(size_t)v * NN] = (_Float16)acc[v];
    }
}

// ---------------------------------------------------------------------------
// Softmax row stats of S2 = (log2e*f)^T g (exp2 domain): one wave per 16 rows,
// stream 256 col-tiles (one K=32 WMMA each), online (max, rescaled sum),
// 16-lane shuffle merge.  Mrow = rowmax(S2), invZ = 1/sum(2^(S2-M)).
// ---------------------------------------------------------------------------
__global__ void __launch_bounds__(256) softmax_stats(
    const _Float16* __restrict__ fT, const _Float16* __restrict__ gT,
    float* __restrict__ Mrow, float* __restrict__ invZ)
{
    const int wave = blockIdx.x * 8 + (threadIdx.x >> 5);
    if (wave >= BB * (NN / 16)) return;                   // wave-uniform
    const int b  = wave / (NN / 16);
    const int k0 = (wave % (NN / 16)) * 16;
    const int l = threadIdx.x & 31, col = l & 15, half = l >> 4;

    const v16h a = load_fragA(fT + ((size_t)b * NN + k0 + col) * C8V, half);
    const _Float16* gbase = gT + ((size_t)b * NN + col) * C8V;

    float m[8], z[8];
    #pragma unroll
    for (int v = 0; v < 8; ++v) { m[v] = -3.0e38f; z[v] = 0.0f; }

    #pragma unroll 4
    for (int j0 = 0; j0 < NN; j0 += 16) {
        v8f s = {};
        s = wmma_f16(a, load_fragB(gbase + (size_t)j0 * C8V, half), s);
        #pragma unroll
        for (int v = 0; v < 8; ++v) {
            const float sv = s[v];
            const float mn = fmaxf(m[v], sv);
            z[v] = z[v] * exp2fast(m[v] - mn) + exp2fast(sv - mn);
            m[v] = mn;
        }
    }
    #pragma unroll
    for (int mask = 1; mask < 16; mask <<= 1) {
        #pragma unroll
        for (int v = 0; v < 8; ++v) {
            const float mo = __shfl_xor(m[v], mask, 32);
            const float zo = __shfl_xor(z[v], mask, 32);
            const float mn = fmaxf(m[v], mo);
            z[v] = z[v] * exp2fast(m[v] - mn) + zo * exp2fast(mo - mn);
            m[v] = mn;
        }
    }
    if (col == 0) {
        #pragma unroll
        for (int v = 0; v < 8; ++v) {
            const int k = k0 + v + 8 * half;
            Mrow[(size_t)b * NN + k] = m[v];
            invZ[(size_t)b * NN + k] = 1.0f / z[v];
        }
    }
}

// ---------------------------------------------------------------------------
// Fused attention-apply. Block = (b, 64 output cols). Per 64-k block:
//   phase 1: recompute S2 (16 WMMAs across 8 waves), E' = 2^(S2-M)*invZ,
//            staged TRANSPOSED [i][k] in a DOUBLE-BUFFERED LDS tile
//            (one barrier per k-block; phase-2 reads overlap next phase-1).
//   phase 2: per wave 16 WMMAs: acc(32c x 64i) += hh(A, global b128) @ E'(LDS b128).
// Epilogue: out = gamma*sa + x (coalesced dword ops).
// ---------------------------------------------------------------------------
__global__ void __launch_bounds__(256) attn_out(
    const _Float16* __restrict__ fT, const _Float16* __restrict__ gT,
    const _Float16* __restrict__ hh, const float* __restrict__ Mrow,
    const float* __restrict__ invZ, const float* __restrict__ x,
    const float* __restrict__ gamma, float* __restrict__ out)
{
    __shared__ _Float16 E[2][64 * ETS];                   // E'[i][k], 2 x 10 KB
    const int b  = blockIdx.y;
    const int i0 = blockIdx.x * 64;
    const int w  = threadIdx.x >> 5, l = threadIdx.x & 31;
    const int col = l & 15, half = l >> 4;
    const int crow0 = w * 32;

    const _Float16* fbase = fT + ((size_t)b * NN + col) * C8V;
    const float*    Mb    = Mrow + (size_t)b * NN;
    const float*    Zb    = invZ + (size_t)b * NN;
    const _Float16* hrow0 = hh + ((size_t)b * CC + crow0 + col) * NN;
    const _Float16* hrow1 = hrow0 + (size_t)16 * NN;
    const float g0 = gamma[0];

    const v8f vzero = {};
    v8f acc[2][4];
    #pragma unroll
    for (int rt = 0; rt < 2; ++rt)
        #pragma unroll
        for (int ct = 0; ct < 4; ++ct) acc[rt][ct] = vzero;

    for (int k0 = 0; k0 < NN; k0 += 64) {
        _Float16* Eb = E[(k0 >> 6) & 1];
        if (k0 + 64 < NN) {                               // global_prefetch_b8
            __builtin_prefetch(hrow0 + k0 + 64, 0, 1);
            __builtin_prefetch(hrow1 + k0 + 64, 0, 1);
        }
        // ---- phase 1: E' tile (transposed) into current LDS buffer ----
        #pragma unroll
        for (int t = 0; t < 2; ++t) {
            const int idx = w * 2 + t;                    // 16 tiles / 8 waves
            const int tk = idx >> 2, ti = idx & 3;
            v16h a  = load_fragA(fbase + (size_t)(k0 + tk * 16) * C8V, half);
            v16h bb = load_fragB(gT + ((size_t)b * NN + i0 + ti * 16 + col) * C8V, half);
            v8f s = {};
            s = wmma_f16(a, bb, s);
            const int kr = k0 + tk * 16 + 8 * half;
            const v8f mv = *(const v8f*)(Mb + kr);        // 8 consecutive rows
            const v8f zv = *(const v8f*)(Zb + kr);
            v8h eh;
            #pragma unroll
            for (int v = 0; v < 8; ++v)
                eh[v] = (_Float16)(exp2fast(s[v] - mv[v]) * zv[v]);
            *(v8h*)(&Eb[(ti * 16 + col) * ETS + tk * 16 + 8 * half]) = eh;
        }
        __syncthreads();   // writes of buf p visible; buf p overwritten only
                           // after the NEXT barrier -> no trailing barrier
        // ---- phase 2 ----
        v16h afr[2][2];
        #pragma unroll
        for (int kk = 0; kk < 2; ++kk) {
            afr[0][kk] = load_fragA(hrow0 + k0 + kk * 32, half);
            afr[1][kk] = load_fragA(hrow1 + k0 + kk * 32, half);
        }
        #pragma unroll
        for (int ct = 0; ct < 4; ++ct) {
            #pragma unroll
            for (int kk = 0; kk < 2; ++kk) {
                const v16h bb = load_fragB(&Eb[(ct * 16 + col) * ETS + kk * 32], half);
                #pragma unroll
                for (int rt = 0; rt < 2; ++rt)
                    acc[rt][ct] = wmma_f16(afr[rt][kk], bb, acc[rt][ct]);
            }
        }
    }

    // ---- epilogue ----
    const size_t obase = ((size_t)b * CC + crow0 + 8 * half) * NN + i0 + col;
    #pragma unroll
    for (int rt = 0; rt < 2; ++rt)
        #pragma unroll
        for (int ct = 0; ct < 4; ++ct)
            #pragma unroll
            for (int v = 0; v < 8; ++v) {
                const size_t idx = obase + ((size_t)rt * 16 + v) * NN + ct * 16;
                out[idx] = g0 * acc[rt][ct][v] + x[idx];
            }
}

// ---------------------------------------------------------------------------
extern "C" void kernel_launch(void* const* d_in, const int* in_sizes, int n_in,
                              void* d_out, int out_size, void* d_ws, size_t ws_size,
                              hipStream_t stream) {
    (void)in_sizes; (void)n_in; (void)out_size; (void)ws_size;
    const float* x     = (const float*)d_in[0];
    const float* Wf    = (const float*)d_in[1];
    const float* bf    = (const float*)d_in[2];
    const float* Wg    = (const float*)d_in[3];
    const float* bg    = (const float*)d_in[4];
    const float* Wh    = (const float*)d_in[5];
    const float* bh    = (const float*)d_in[6];
    const float* gamma = (const float*)d_in[7];
    float* out = (float*)d_out;

    char* ws = (char*)d_ws;
    _Float16* xT   = (_Float16*)ws; ws += (size_t)BB * NN * CC  * sizeof(_Float16);
    _Float16* fT   = (_Float16*)ws; ws += (size_t)BB * NN * C8V * sizeof(_Float16);
    _Float16* gTt  = (_Float16*)ws; ws += (size_t)BB * NN * C8V * sizeof(_Float16);
    _Float16* hhh  = (_Float16*)ws; ws += (size_t)BB * CC * NN  * sizeof(_Float16);
    _Float16* Wf_h = (_Float16*)ws; ws += (size_t)C8V * CC * sizeof(_Float16);
    _Float16* Wg_h = (_Float16*)ws; ws += (size_t)C8V * CC * sizeof(_Float16);
    _Float16* Wh_h = (_Float16*)ws; ws += (size_t)CC  * CC * sizeof(_Float16);
    float* Mrow = (float*)ws; ws += (size_t)BB * NN * sizeof(float);
    float* invZ = (float*)ws; ws += (size_t)BB * NN * sizeof(float);

    {   // converts (one-time)
        dim3 g(NN / 64, CC / 32, BB);
        x_to_f16T<<<g, 256, 0, stream>>>(x, xT);
        f32_to_f16<<<(C8V * CC + 255) / 256, 256, 0, stream>>>(Wf, Wf_h, C8V * CC);
        f32_to_f16<<<(C8V * CC + 255) / 256, 256, 0, stream>>>(Wg, Wg_h, C8V * CC);
        f32_to_f16<<<(CC * CC + 255) / 256, 256, 0, stream>>>(Wh, Wh_h, CC * CC);
    }
    {   // f (pre-scaled by log2e), g -> transposed (B,N,32) f16
        const int tiles  = BB * (C8V / 16) * (NN / 16);
        const int blocks = (tiles + 7) / 8;
        proj_gemm<1><<<blocks, 256, 0, stream>>>(Wf_h, bf, xT, C8V, LOG2E, fT);
        proj_gemm<1><<<blocks, 256, 0, stream>>>(Wg_h, bg, xT, C8V, 1.0f, gTt);
    }
    {   // hh -> (B,256,N) f16
        const int tiles  = BB * (CC / 16) * (NN / 16);
        const int blocks = (tiles + 7) / 8;
        proj_gemm<0><<<blocks, 256, 0, stream>>>(Wh_h, bh, xT, CC, 1.0f, hhh);
    }
    {   // softmax row stats (exp2 domain)
        const int waves = BB * (NN / 16);
        softmax_stats<<<(waves + 7) / 8, 256, 0, stream>>>(fT, gTt, Mrow, invZ);
    }
    {   // fused attention apply
        dim3 grid(NN / 64, BB);
        attn_out<<<grid, 256, 0, stream>>>(fT, gTt, hhh, Mrow, invZ, x, gamma, out);
    }
}
